// End2EndBiLSTMModel_47828755808359
// MI455X (gfx1250) — matrix operand order repaired and checked by reading, not compiled, not run on hardware
//
#include <hip/hip_runtime.h>
#include <hip/hip_bf16.h>

// ---------------------------------------------------------------------------
// Problem constants (match reference)
// ---------------------------------------------------------------------------
constexpr int kT   = 512;
constexpr int kB   = 64;
constexpr int kDin = 256;
constexpr int kH   = 512;
constexpr int kOut = 128;
constexpr int kG   = 4 * kH;      // 2048 gate columns
constexpr int kNWG = 16;          // workgroups per direction in the scan
constexpr float kClip = 100.0f;

// ---------------------------------------------------------------------------
// Types
// ---------------------------------------------------------------------------
typedef __bf16 bf16;
typedef __attribute__((ext_vector_type(16))) __bf16 v16bf;
typedef __attribute__((ext_vector_type(8)))  __bf16 v8bf;
typedef __attribute__((ext_vector_type(8)))  float  v8f;
typedef __attribute__((ext_vector_type(4)))  unsigned int uint4v;
typedef __attribute__((ext_vector_type(8)))  int int8v;
typedef __attribute__((ext_vector_type(4)))  int int4v;
typedef int v4i_g __attribute__((vector_size(16)));  // gcc-style, for builtins

#define AS1 __attribute__((address_space(1)))
#define AS3 __attribute__((address_space(3)))

// CDNA5 feature detection (fallbacks keep both toolchains compiling)
#if defined(__HIP_DEVICE_COMPILE__) &&                                     \
    __has_builtin(__builtin_amdgcn_global_load_async_to_lds_b128) &&       \
    __has_builtin(__builtin_amdgcn_s_wait_asynccnt)
#define USE_ASYNC_LDS 1
#else
#define USE_ASYNC_LDS 0
#endif

#if defined(__HIP_DEVICE_COMPILE__) &&                                     \
    __has_builtin(__builtin_amdgcn_tensor_load_to_lds) &&                  \
    __has_builtin(__builtin_amdgcn_s_wait_tensorcnt)
#define USE_TDM 1
#else
#define USE_TDM 0
#endif

// 16B async global->LDS copy (ASYNCcnt) with synchronous fallback.
// Builtin signature (from hipcc diagnostic): (v4i AS1*, v4i AS3*, Iimm, Iimm).
__device__ inline void async_g2l_b128(const void* g, void* l) {
#if USE_ASYNC_LDS
  __builtin_amdgcn_global_load_async_to_lds_b128(
      (AS1 v4i_g*)g, (AS3 v4i_g*)l, 0, 0);
#else
  *(v8bf*)l = *(const v8bf*)g;
#endif
}

template <unsigned short N>
__device__ inline void wait_async() {
#if USE_ASYNC_LDS
  __builtin_amdgcn_s_wait_asynccnt(N);
#endif
}

// ---------------------------------------------------------------------------
// WMMA helpers (wave32 v_wmma_f32_16x16x32_bf16)
// ---------------------------------------------------------------------------
__device__ inline v8f wmma_bf16(v16bf a, v16bf b, v8f c) {
  return __builtin_amdgcn_wmma_f32_16x16x32_bf16(false, a, false, b,
                                                 (short)0, c, false, false);
}

// A fragment: 16x32 bf16 tile of row-major A(ld). ISA 7.12.2 layout.
__device__ inline v16bf load_frag_a(const bf16* base, int ld, int m0, int k0,
                                    int lane) {
  const int r = lane & 15, hi = lane >> 4;
  const bf16* p = base + (size_t)(m0 + r) * ld + (k0 + hi * 8);
  v8bf lo = *(const v8bf*)p;
  v8bf hv = *(const v8bf*)(p + 16);
  return __builtin_shufflevector(lo, hv, 0, 1, 2, 3, 4, 5, 6, 7,
                                 8, 9, 10, 11, 12, 13, 14, 15);
}

// B fragment: B[k][n] = W[n0+n][k] with W row-major (N,K): per lane two
// 8-bf16 runs from W's row at k0 + (lane>>4)*8 and k0 + 16 + (lane>>4)*8.
__device__ inline v16bf load_frag_b(const bf16* base, int ld, int n0, int k0,
                                    int lane) {
  const int n = lane & 15, hi = lane >> 4;
  const bf16* p = base + (size_t)(n0 + n) * ld + (k0 + hi * 8);
  v8bf lo = *(const v8bf*)p;
  v8bf hv = *(const v8bf*)(p + 16);
  return __builtin_shufflevector(lo, hv, 0, 1, 2, 3, 4, 5, 6, 7,
                                 8, 9, 10, 11, 12, 13, 14, 15);
}

__device__ inline float sigmoidf_(float x) { return 1.f / (1.f + __expf(-x)); }

// ---------------------------------------------------------------------------
// fp32 -> bf16 conversion (grid-stride)
// ---------------------------------------------------------------------------
__global__ void k_cvt(const float* __restrict__ src, bf16* __restrict__ dst,
                      size_t n) {
  size_t i = (size_t)blockIdx.x * blockDim.x + threadIdx.x;
  size_t stride = (size_t)gridDim.x * blockDim.x;
  for (; i < n; i += stride) dst[i] = (bf16)src[i];
}

__global__ void k_zero_u32(unsigned* p, int n) {
  int i = threadIdx.x;
  if (i < n) p[i] = 0u;
}

// ---------------------------------------------------------------------------
// Input-projection GEMM: C(M,kG) = A(M,K)_bf16 * W(kG,K)_bf16^T + bias
// 256 threads = 8 waves; block tile 128(M) x 64(N); A tile staged in LDS via
// double-buffered async copies; B fragments direct from global (weights are
// L2-resident: 192MB L2 vs <20MB of weights).
// ---------------------------------------------------------------------------
template <int K>
__global__ __launch_bounds__(256) void k_gemm_xw(
    const bf16* __restrict__ A, const bf16* __restrict__ W,
    const float* __restrict__ bias, float* __restrict__ C) {
  __shared__ bf16 sA[2][128 * 32];  // 2 x 8KB double buffer

  const int tid = threadIdx.x;
  const int lane = tid & 31;
  const int wv = tid >> 5;
  const int wm = wv >> 2, wn = wv & 3;
  const int n0 = blockIdx.y * 64 + wn * 16;
  const int mb = blockIdx.x * 128;
  constexpr int NS = K / 32;

  // Stage a 128x32 A tile (512 16B chunks, 2 per thread).
  auto stage = [&](int buf, int k0) {
#pragma unroll
    for (int j = 0; j < 2; ++j) {
      const int c = tid + j * 256;
      const int row = c >> 2, sub = c & 3;
      async_g2l_b128(A + (size_t)(mb + row) * K + k0 + sub * 8,
                     &sA[buf][row * 32 + sub * 8]);
    }
  };

  stage(0, 0);
  v8f acc[4] = {};
  for (int ks = 0; ks < NS; ++ks) {
    if (ks + 1 < NS) {
      stage((ks + 1) & 1, (ks + 1) * 32);
      wait_async<2>();  // previous stage's 2 per-wave async ops done
    } else {
      wait_async<0>();
    }
    __syncthreads();

    const bf16* At = sA[ks & 1];
    v16bf bf = load_frag_b(W, K, n0, ks * 32, lane);
    if (ks + 1 < NS)  // pull next B segment toward the WGP
      __builtin_prefetch(W + (size_t)(n0 + (lane & 15)) * K + (ks + 1) * 32,
                         0, 1);
#pragma unroll
    for (int s = 0; s < 4; ++s) {
      v16bf af = load_frag_a(At, 32, wm * 64 + s * 16, 0, lane);
      acc[s] = wmma_bf16(af, bf, acc[s]);
    }
    __syncthreads();
  }

  const int n = lane & 15, hi = lane >> 4;
  const float bv = bias[n0 + n];
#pragma unroll
  for (int s = 0; s < 4; ++s)
#pragma unroll
    for (int i = 0; i < 8; ++i)
      C[(size_t)(mb + wm * 64 + s * 16 + hi * 8 + i) * kG + n0 + n] =
          acc[s][i] + bv;
}

// ---------------------------------------------------------------------------
// Persistent bidirectional LSTM scan.
// grid = (kNWG, 2 dirs), block = 256 (8 waves).
// WG wg owns hidden units [wg*32, wg*32+32): all 4 gate blocks for those
// units, so the nonlinearity needs only a WG-local barrier.
// LDS: Whh slice 128KB (TDM-preloaded) + h state 64KB (async-staged per step)
//      + gates 32KB = 224KB of the 320KB WGP LDS.
// Cross-WG sync per step: monotonic atomic counter + s_sleep spin.
// ---------------------------------------------------------------------------
__global__ __launch_bounds__(256) void k_scan(
    const float* __restrict__ xw_base,   // [2][T*B*kG] fp32 (bias folded in)
    const bf16* __restrict__ whh_base,   // [2][kG*kH]  bf16
    bf16* __restrict__ hstate_base,      // [2][B*kH]   bf16
    bf16* __restrict__ out_cat,          // [T*B*2H] bf16 or null (layer0)
    float* __restrict__ out_seq_base,    // [2][T*B*H] fp32 or null (layer1)
    unsigned* __restrict__ sync) {       // one counter per dir
  extern __shared__ char smem[];
  bf16* lds_whh = (bf16*)smem;                                 // 128 x kH
  bf16* lds_h   = (bf16*)(smem + 128 * kH * 2);                // kB  x kH
  float* lds_g  = (float*)(smem + 128 * kH * 2 + kB * kH * 2); // kB x 128

  const int dir = blockIdx.y;
  const int wg = blockIdx.x;
  const int tid = threadIdx.x;
  const int lane = tid & 31, wv = tid >> 5;
  const int j0 = wg * 32;

  const float* xw = xw_base + (size_t)dir * kT * kB * kG;
  const bf16* whh = whh_base + (size_t)dir * kG * kH;
  bf16* hstate = hstate_base + (size_t)dir * kB * kH;
  float* out_seq =
      out_seq_base ? out_seq_base + (size_t)dir * kT * kB * kH : nullptr;
  unsigned* cnt = sync + dir;

  // ---- Preload this WG's Whh slice: 4 tiles of 32 rows x 512 cols bf16. ----
#if USE_TDM
  if (tid < 32) {  // wave 0 drives the Tensor Data Mover
    const unsigned lds0 =
        (unsigned)(unsigned long long)(AS3 void*)(void*)lds_whh;
#pragma unroll
    for (int g = 0; g < 4; ++g) {
      const unsigned long long ga =
          (unsigned long long)(whh + (size_t)(g * kH + j0) * kH);
      uint4v g0;
      g0[0] = 1u;                                     // count=1, load
      g0[1] = lds0 + (unsigned)g * (32u * kH * 2u);   // lds_addr
      g0[2] = (unsigned)(ga & 0xFFFFFFFFull);         // global_addr[31:0]
      g0[3] = (unsigned)((ga >> 32) & 0x01FFFFFFull)  // global_addr[56:32]
              | (2u << 30);                           // type = 2 ("image")
      int8v g1 = {};
      g1[0] = (int)(1u << 16);                  // data_size = 2 bytes
      g1[1] = (int)(((unsigned)kH) << 16);      // tensor_dim0 = 512
      g1[2] = (int)(32u << 16);                 // tensor_dim1 = 32
      g1[3] = (int)(((unsigned)kH) << 16);      // tile_dim0 = 512
      g1[4] = 32;                               // tile_dim1 = 32
      g1[5] = kH;                               // tensor_dim0_stride = 512
      const int4v z4 = {};
#if __clang_major__ >= 23
      const int8v z8 = {};
      __builtin_amdgcn_tensor_load_to_lds(g0, g1, z4, z4, z8, 0);
#else
      __builtin_amdgcn_tensor_load_to_lds(g0, g1, z4, z4, 0);
#endif
    }
    __builtin_amdgcn_s_wait_tensorcnt(0);
  }
#else
  for (int idx = tid; idx < 128 * (kH / 8); idx += 256) {
    const int lr = idx / (kH / 8), ch = idx % (kH / 8);
    const int g = lr >> 5, jj = lr & 31;
    *((v8bf*)(lds_whh + (size_t)lr * kH) + ch) =
        *((const v8bf*)(whh + (size_t)(g * kH + j0 + jj) * kH) + ch);
  }
#endif
  __syncthreads();

  float creg[8];
#pragma unroll
  for (int u = 0; u < 8; ++u) creg[u] = 0.f;

  for (int t = 0; t < kT; ++t) {
    const int tt = dir ? (kT - 1 - t) : t;

    if (t > 0) {
      // Stage h(t-1) into LDS via async copies (16 per-wave ops).
      for (int idx = tid; idx < kB * (kH / 8); idx += 256)
        async_g2l_b128((const v8bf*)hstate + idx, (v8bf*)lds_h + idx);
      wait_async<0>();
      __syncthreads();

      // gates(64 x 128) += h(64 x 512) * Whh_slice^T
      v8f acc[4] = {};
#pragma unroll 4
      for (int k0 = 0; k0 < kH; k0 += 32) {
        v16bf bf = load_frag_b(lds_whh, kH, wv * 16, k0, lane);
#pragma unroll
        for (int s = 0; s < 4; ++s) {
          v16bf af = load_frag_a(lds_h, kH, s * 16, k0, lane);
          acc[s] = wmma_bf16(af, bf, acc[s]);
        }
      }
      const int n = lane & 15, hi = lane >> 4;
#pragma unroll
      for (int s = 0; s < 4; ++s)
#pragma unroll
        for (int i = 0; i < 8; ++i)
          lds_g[(s * 16 + hi * 8 + i) * 128 + wv * 16 + n] = acc[s][i];
    }
    __syncthreads();

    // Nonlinearities + state update; cell state lives in registers.
    const int tnext = dir ? (tt - 1) : (tt + 1);
#pragma unroll
    for (int u = 0; u < 8; ++u) {
      const int e = u * 256 + tid;  // 0..2047
      const int b = e >> 5, jj = e & 31;
      const int j = j0 + jj;
      const float* xr = xw + ((size_t)tt * kB + b) * kG;
      if (tnext >= 0 && tnext < kT)  // distance-1 prefetch of next xW rows
        __builtin_prefetch(xw + ((size_t)tnext * kB + b) * kG + j, 0, 1);
      float gi = xr[j];
      float gf = xr[kH + j];
      float gg = xr[2 * kH + j];
      float go = xr[3 * kH + j];
      if (t > 0) {
        gi += lds_g[b * 128 + jj];
        gf += lds_g[b * 128 + 32 + jj];
        gg += lds_g[b * 128 + 64 + jj];
        go += lds_g[b * 128 + 96 + jj];
      }
      const float i_ = sigmoidf_(gi);
      const float f_ = sigmoidf_(gf);
      const float g_ = tanhf(gg);
      const float o_ = sigmoidf_(go);
      float c = f_ * creg[u] + i_ * g_;
      c = fminf(fmaxf(c, -kClip), kClip);
      creg[u] = c;
      const float h = o_ * tanhf(c);
      hstate[b * kH + j] = (bf16)h;
      if (out_cat)
        out_cat[(size_t)tt * kB * (2 * kH) + b * (2 * kH) + dir * kH + j] =
            (bf16)h;
      if (out_seq) out_seq[(size_t)tt * kB * kH + b * kH + j] = h;
    }

    // Per-direction grid barrier: h(t) visible before t+1 GEMM.
    __syncthreads();
    if (tid == 0) {
      __threadfence();
      atomicAdd(cnt, 1u);
      const unsigned want = (unsigned)(t + 1) * kNWG;
      volatile unsigned* vc = (volatile unsigned*)cnt;
      while (*vc < want) __builtin_amdgcn_s_sleep(1);
    }
    __syncthreads();
    __threadfence();
  }
}

// ---------------------------------------------------------------------------
// Attention: one block per batch element.
// ---------------------------------------------------------------------------
__global__ __launch_bounds__(256) void k_attn(
    const float* __restrict__ hf, const float* __restrict__ hb,
    const float* __restrict__ Wa, const float* __restrict__ ba,
    float* __restrict__ attn_out /* [B*T] */,
    float* __restrict__ pooled /* [B*H] */) {
  const int b = blockIdx.x, tid = threadIdx.x;
  __shared__ float logits[kT];
  __shared__ float red[256];

  for (int t = tid; t < kT; t += 256) {
    const float* pf = hf + ((size_t)t * kB + b) * kH;
    const float* pb = hb + ((size_t)t * kB + b) * kH;
    float s = 0.f;
    for (int h = 0; h < kH; ++h) s += (pf[h] + pb[h]) * Wa[h];
    logits[t] = s + ba[0];
  }
  __syncthreads();

  float m = -3.4e38f;
  for (int t = tid; t < kT; t += 256) m = fmaxf(m, logits[t]);
  red[tid] = m;
  __syncthreads();
  for (int s = 128; s > 0; s >>= 1) {
    if (tid < s) red[tid] = fmaxf(red[tid], red[tid + s]);
    __syncthreads();
  }
  m = red[0];
  __syncthreads();

  float ssum = 0.f;
  for (int t = tid; t < kT; t += 256) {
    const float e = __expf(logits[t] - m);
    logits[t] = e;
    ssum += e;
  }
  red[tid] = ssum;
  __syncthreads();
  for (int s = 128; s > 0; s >>= 1) {
    if (tid < s) red[tid] += red[tid + s];
    __syncthreads();
  }
  const float inv = 1.f / red[0];
  __syncthreads();

  for (int t = tid; t < kT; t += 256) {
    const float a = logits[t] * inv;
    logits[t] = a;
    attn_out[(size_t)b * kT + t] = a;
  }
  __syncthreads();

  for (int h = tid; h < kH; h += 256) {
    float acc = 0.f;
    for (int t = 0; t < kT; ++t)
      acc += logits[t] * (hf[((size_t)t * kB + b) * kH + h] +
                          hb[((size_t)t * kB + b) * kH + h]);
    pooled[(size_t)b * kH + h] = acc;
  }
}

__global__ void k_pred(const float* __restrict__ pooled,
                       const float* __restrict__ Wo,
                       const float* __restrict__ bo, float* __restrict__ pred) {
  const int idx = blockIdx.x * blockDim.x + threadIdx.x;
  if (idx >= kB * kOut) return;
  const int b = idx / kOut, o = idx % kOut;
  float s = bo[o];
  const float* pp = pooled + (size_t)b * kH;
  const float* pw = Wo + (size_t)o * kH;
  for (int h = 0; h < kH; ++h) s += pp[h] * pw[h];
  pred[idx] = s;
}

// ---------------------------------------------------------------------------
// Host launcher
// ---------------------------------------------------------------------------
extern "C" void kernel_launch(void* const* d_in, const int* in_sizes, int n_in,
                              void* d_out, int out_size, void* d_ws,
                              size_t ws_size, hipStream_t stream) {
  (void)in_sizes; (void)n_in; (void)out_size; (void)ws_size;
  const float* x    = (const float*)d_in[0];
  const float* Wih0 = (const float*)d_in[1];
  const float* Whh0 = (const float*)d_in[2];
  const float* b0   = (const float*)d_in[3];
  const float* Wih1 = (const float*)d_in[4];
  const float* Whh1 = (const float*)d_in[5];
  const float* b1   = (const float*)d_in[6];
  const float* Wa   = (const float*)d_in[7];
  const float* ba   = (const float*)d_in[8];
  const float* Wo   = (const float*)d_in[9];
  const float* bo   = (const float*)d_in[10];

  char* ws = (char*)d_ws;
  size_t off = 0;
  auto take = [&](size_t bytes) -> char* {
    char* p = ws + off;
    off = (off + bytes + 255) & ~(size_t)255;
    return p;
  };

  bf16* x_b     = (bf16*)take((size_t)kT * kB * kDin * sizeof(bf16));
  bf16* wih0_b  = (bf16*)take((size_t)2 * kG * kDin * sizeof(bf16));
  bf16* whh0_b  = (bf16*)take((size_t)2 * kG * kH * sizeof(bf16));
  bf16* wih1_b  = (bf16*)take((size_t)2 * kG * (2 * kH) * sizeof(bf16));
  bf16* whh1_b  = (bf16*)take((size_t)2 * kG * kH * sizeof(bf16));
  float* xw     = (float*)take((size_t)2 * kT * kB * kG * sizeof(float));
  bf16* hcat0   = (bf16*)take((size_t)kT * kB * (2 * kH) * sizeof(bf16));
  float* hseq1  = (float*)take((size_t)2 * kT * kB * kH * sizeof(float));
  bf16* hstate  = (bf16*)take((size_t)2 * kB * kH * sizeof(bf16));
  float* pooled = (float*)take((size_t)kB * kH * sizeof(float));
  unsigned* sync = (unsigned*)take(256);

  // --- fp32 -> bf16 operands ---
  k_cvt<<<512, 256, 0, stream>>>(x, x_b, (size_t)kT * kB * kDin);
  k_cvt<<<256, 256, 0, stream>>>(Wih0, wih0_b, (size_t)2 * kG * kDin);
  k_cvt<<<256, 256, 0, stream>>>(Whh0, whh0_b, (size_t)2 * kG * kH);
  k_cvt<<<256, 256, 0, stream>>>(Wih1, wih1_b, (size_t)2 * kG * (2 * kH));
  k_cvt<<<256, 256, 0, stream>>>(Whh1, whh1_b, (size_t)2 * kG * kH);

  const dim3 ggrid((kT * kB) / 128, kG / 64);
  const size_t shmem = (size_t)128 * kH * 2 + (size_t)kB * kH * 2 +
                       (size_t)kB * 128 * 4;  // 224 KB

  // --- layer 0: input projections then scan ---
  for (int d = 0; d < 2; ++d)
    k_gemm_xw<kDin><<<ggrid, 256, 0, stream>>>(
        x_b, wih0_b + (size_t)d * kG * kDin, b0 + (size_t)d * kG,
        xw + (size_t)d * kT * kB * kG);
  k_zero_u32<<<1, 32, 0, stream>>>(sync, 8);
  k_scan<<<dim3(kNWG, 2), 256, shmem, stream>>>(xw, whh0_b, hstate, hcat0,
                                                nullptr, sync);

  // --- layer 1: input projections (K = 2H) then scan ---
  for (int d = 0; d < 2; ++d)
    k_gemm_xw<2 * kH><<<ggrid, 256, 0, stream>>>(
        hcat0, wih1_b + (size_t)d * kG * (2 * kH), b1 + (size_t)d * kG,
        xw + (size_t)d * kT * kB * kG);
  k_zero_u32<<<1, 32, 0, stream>>>(sync, 8);
  k_scan<<<dim3(kNWG, 2), 256, shmem, stream>>>(xw, whh1_b, hstate, nullptr,
                                                hseq1, sync);

  // --- attention pooling + head (d_out = [pred | attn]) ---
  float* pred = (float*)d_out;
  float* attn = (float*)d_out + (size_t)kB * kOut;
  k_attn<<<kB, 256, 0, stream>>>(hseq1, hseq1 + (size_t)kT * kB * kH, Wa, ba,
                                 attn, pooled);
  k_pred<<<(kB * kOut + 255) / 256, 256, 0, stream>>>(pooled, Wo, bo, pred);
}
